// LTFAD_84104049590414
// MI455X (gfx1250) — matrix operand order repaired and checked by reading, not compiled, not run on hardware
//
#include <hip/hip_runtime.h>
#include <math.h>

typedef __attribute__((ext_vector_type(16))) _Float16 v16h;
typedef __attribute__((ext_vector_type(8)))  _Float16 v8h;
typedef __attribute__((ext_vector_type(8)))  float    v8f;

#define LAMBD_F 0.01f

// ---------------------------------------------------------------- small utils

__global__ void cvt_f32_f16(const float* __restrict__ s, _Float16* __restrict__ d, int n) {
  for (int i = blockIdx.x * blockDim.x + threadIdx.x; i < n; i += gridDim.x * blockDim.x)
    d[i] = (_Float16)s[i];
}

// dst[0:n] = a, dst[n:2n] = b  (f32)
__global__ void cat2_f32(float* __restrict__ dst, const float* __restrict__ a,
                         const float* __restrict__ b, int n) {
  for (int i = blockIdx.x * blockDim.x + threadIdx.x; i < 2 * n; i += gridDim.x * blockDim.x)
    dst[i] = (i < n) ? a[i] : b[i - n];
}

// Combined complex layer-1 weight: [2d x 512] = [[r, i], [-i, r]] (r,i are [d x 256])
__global__ void build_w1c(_Float16* __restrict__ dst, const float* __restrict__ r,
                          const float* __restrict__ im, int d) {
  int n = 2 * d * 512;
  for (int idx = blockIdx.x * blockDim.x + threadIdx.x; idx < n; idx += gridDim.x * blockDim.x) {
    int k = idx >> 9, j = idx & 511;
    float v;
    if (j < 256) v = (k < d) ? r[k * 256 + j] : -im[(k - d) * 256 + j];
    else {
      int jj = j - 256;
      v = (k < d) ? im[k * 256 + jj] : r[(k - d) * 256 + jj];
    }
    dst[idx] = (_Float16)v;
  }
}

// Combined complex layer-2 weight: [512 x 2g] = [[r, i], [-i, r]] (r,i are [256 x g])
__global__ void build_w2c(_Float16* __restrict__ dst, const float* __restrict__ r,
                          const float* __restrict__ im, int g) {
  int n = 512 * 2 * g;
  for (int idx = blockIdx.x * blockDim.x + threadIdx.x; idx < n; idx += gridDim.x * blockDim.x) {
    int k = idx / (2 * g), j = idx % (2 * g);
    float v;
    if (j < g) v = (k < 256) ? r[k * g + j] : -im[(k - 256) * g + j];
    else {
      int jj = j - g;
      v = (k < 256) ? im[k * g + jj] : r[(k - 256) * g + jj];
    }
    dst[idx] = (_Float16)v;
  }
}

// Forward DFT (ortho rfft) matrices Cc/Cs [288 x 512] (rows f>=257 zero) and
// inverse (ortho irfft) matrices Ir/Ii [512 x 288] (cols f>=257 zero, alpha folded in).
__global__ void gen_dft(_Float16* __restrict__ Cc, _Float16* __restrict__ Cs,
                        _Float16* __restrict__ Ir, _Float16* __restrict__ Ii) {
  const float inv = 0.04419417382415922f;   // 1/sqrt(512)
  const float w   = 0.01227184630308513f;   // 2*pi/512
  for (int i = blockIdx.x * blockDim.x + threadIdx.x; i < 288 * 512; i += gridDim.x * blockDim.x) {
    {
      int f = i >> 9, t = i & 511;
      float cr = 0.f, sr = 0.f;
      if (f < 257) {
        int rr = (f * t) & 511;
        float ang = w * (float)rr;
        cr = cosf(ang) * inv;
        sr = -sinf(ang) * inv;
      }
      Cc[i] = (_Float16)cr;
      Cs[i] = (_Float16)sr;
    }
    {
      int t = i / 288, f = i % 288;
      float vr = 0.f, vi = 0.f;
      if (f < 257) {
        float alpha = (f == 0 || f == 256) ? 1.f : 2.f;
        int rr = (f * t) & 511;
        float ang = w * (float)rr;
        vr = alpha * cosf(ang) * inv;
        vi = -alpha * sinf(ang) * inv;
      }
      Ir[i] = (_Float16)vr;
      Ii[i] = (_Float16)vi;
    }
  }
}

// ---------------------------------------------------------------- WMMA GEMM
// D[Mr x N] = sum over 1..2 pairs of A[Mr x K] @ B[K x N], f16 in, f32 accum.
// Tiles are staged into LDS in WMMA *fragment order* so a fragment load is one
// 32-byte contiguous LDS read per lane (ds_load_b128 x2).
// Hot path (aMode==0, K%32==0, lda%8==0): A tile is copied global->LDS with
// gfx1250 async copies (global_load_async_to_lds_b128, ASYNCcnt) straight into
// fragment order; source row is clamped (A row r only feeds guarded out row r).
//
// A fragment map (16x32 f16): lane = (row&15) + 16*((k>>3)&1),
//                             e = (k - 8*hi) < 8 ? (k-8*hi) : (k-8*hi) - 8
// B fragment map (32x16 f16): lane = (j&15) + 16*((k>>3)&1), e = (k>>4)*8 + (k&7)
//
// aMode 1: A element (row=(f,m), col=k) gathered from interleaved re/im buffer
//          X[((f*2 + (k>=aD))*aM + m)*aD + (k % aD)]
// opMode: 0 none; 1 +bias; 2 relu(+bias); 3 shrink(relu(+bias))*scale
// stMode: 0 f32 row-major ldc; 1 f16 row-major ldc;
//         2 f16 scatter (row=(f,m), col j -> Y[((f*2+(j>=sD))*sM+m)*sD + j%sD])
__global__ __launch_bounds__(128)
void gemm_wmma(const _Float16* __restrict__ A, int lda, int aMode, int aM, int aD,
               const _Float16* __restrict__ B, int ldb,
               const _Float16* __restrict__ A2, const _Float16* __restrict__ B2,
               const float* __restrict__ bias,
               float* __restrict__ outF, _Float16* __restrict__ outH, int ldc,
               int opMode, int stMode, int sM, int sD, float scale,
               int Mr, int N, int K) {
  __shared__ v16h sAf[4][32];   // 4 row-subtiles, fragment order
  __shared__ v16h sBf[4][32];   // 4 col-subtiles, fragment order
  const int tid = threadIdx.x;
  const int wave = tid >> 5, lane = tid & 31;
  const int hi = lane >> 4, rl = lane & 15;
  const int rowBlk = blockIdx.y * 64, colBlk = blockIdx.x * 64;

  v8f acc[4] = {};

  const bool fastA = (aMode == 0) && ((K & 31) == 0) && ((lda & 7) == 0);
  const int pairs = (A2 != nullptr) ? 2 : 1;
  for (int pr = 0; pr < pairs; ++pr) {
    const _Float16* pA = pr ? A2 : A;
    const _Float16* pB = pr ? B2 : B;
    for (int kt = 0; kt < K; kt += 32) {
      // ---- stage A tile [64 x 32] into fragment order ----
      if (fastA) {
        // async global->LDS, one b128 per 8 halves, lands in fragment order
        #pragma unroll
        for (int it = 0; it < 2; ++it) {
          int idx = tid + it * 128;          // 256 groups of 8 halves
          int r = idx >> 2, g = idx & 3;
          int gr = rowBlk + r;
          if (gr >= Mr) gr = Mr - 1;         // clamp: pad rows are don't-care
          int k0 = g * 8;
          int hi2 = g & 1;                   // (k0>>3)&1
          int e0 = (g >> 1) * 8;
          int lane2 = (r & 15) + 16 * hi2;
          int st = r >> 4;
          unsigned ldsOff = (unsigned)(size_t)
              ((_Float16*)sAf + ((st * 32 + lane2) * 16 + e0));
          unsigned long long gaddr =
              (unsigned long long)(size_t)(pA + (long)gr * lda + kt + k0);
          asm volatile("global_load_async_to_lds_b128 %0, %1, off"
                       :: "v"(ldsOff), "v"(gaddr)
                       : "memory");
        }
      } else {
        for (int idx = tid; idx < 64 * 32; idx += 128) {
          int r = idx >> 5, k = idx & 31;
          int gr = rowBlk + r, gk = kt + k;
          _Float16 v = (_Float16)0.f;
          if (gr < Mr && gk < K) {
            if (aMode == 0) {
              v = pA[(long)gr * lda + gk];
            } else {
              int f = gr / aM, m = gr - f * aM;
              int ri = (gk >= aD) ? 1 : 0;
              int kk = ri ? gk - aD : gk;
              v = pA[((long)(f * 2 + ri) * aM + m) * aD + kk];
            }
          }
          int st = r >> 4;
          int hi2 = (k >> 3) & 1;
          int kp = k - 8 * hi2;
          int e = (kp < 8) ? kp : kp - 8;
          ((_Float16*)sAf)[(st * 32 + (r & 15) + 16 * hi2) * 16 + e] = v;
        }
      }
      // ---- stage B tile [32 x 64] into fragment order ----
      // per pass: 8 K-values for one column j (coalesced across j), one b128 store
      #pragma unroll
      for (int pass = 0; pass < 2; ++pass) {
        int kg = pass * 2 + (tid >> 6);      // 0..3
        int j = tid & 63;
        int k0 = kg * 8;
        int gj = colBlk + j;
        v8h t = {};
        #pragma unroll
        for (int kk = 0; kk < 8; ++kk) {
          int gk = kt + k0 + kk;
          _Float16 v = (_Float16)0.f;
          if (gk < K && gj < N) v = pB[(long)gk * ldb + gj];
          t[kk] = v;
        }
        int st = j >> 4, col = j & 15;
        int hi2 = (k0 >> 3) & 1, e0 = (k0 >> 4) * 8;
        *(v8h*)((_Float16*)sBf + ((st * 32 + col + 16 * hi2) * 16 + e0)) = t;
      }
      if (fastA) asm volatile("s_wait_asynccnt 0x0" ::: "memory");
      __syncthreads();

      // ---- fragments: one contiguous 32B LDS read each ----
      v16h af = sAf[wave][lane];
      #pragma unroll
      for (int t4 = 0; t4 < 4; ++t4) {
        v16h bf = sBf[t4][lane];
        acc[t4] = __builtin_amdgcn_wmma_f32_16x16x32_f16(
            false, af, false, bf, (short)0, acc[t4], false, false);
      }
      __syncthreads();
    }
  }

  // epilogue: C/D layout -> element r of acc: row = r + 8*hi, col = lane&15
  #pragma unroll
  for (int t4 = 0; t4 < 4; ++t4) {
    #pragma unroll
    for (int r = 0; r < 8; ++r) {
      int grow = rowBlk + wave * 16 + r + hi * 8;
      int gcol = colBlk + t4 * 16 + rl;
      if (grow < Mr && gcol < N) {
        float v = acc[t4][r];
        if (opMode >= 1) v += bias[gcol];
        if (opMode >= 2) v = v > 0.f ? v : 0.f;
        if (opMode >= 3) {
          v = (v > LAMBD_F) ? (v - LAMBD_F) : ((v < -LAMBD_F) ? (v + LAMBD_F) : 0.f);
          v *= scale;
        }
        if (stMode == 0) {
          outF[(long)grow * ldc + gcol] = v;
        } else if (stMode == 1) {
          outH[(long)grow * ldc + gcol] = (_Float16)v;
        } else {
          int f = grow / sM, m = grow - f * sM;
          int ri = (gcol >= sD) ? 1 : 0;
          int j = ri ? gcol - sD : gcol;
          outH[((long)(f * 2 + ri) * sM + m) * sD + j] = (_Float16)v;
        }
      }
    }
  }
}

// ---------------------------------------------------------------- host driver

extern "C" void kernel_launch(void* const* d_in, const int* in_sizes, int n_in,
                              void* d_out, int out_size, void* d_ws, size_t ws_size,
                              hipStream_t stream) {
  (void)in_sizes; (void)n_in; (void)out_size; (void)ws_size;
  const int Bb = 4, Ll = 512, Mm = 55, Fp = 288;
  const int ds[3] = {3, 5, 7};
  const float sdrop = 1.0f / 0.95f;

  // --- workspace carve-out ---
  char* base = (char*)d_ws;
  size_t off = 0;
  auto alloc = [&](size_t bytes) -> char* {
    off = (off + 255) & ~(size_t)255;
    char* p = base + off;
    off += bytes;
    return p;
  };

  _Float16* Cc  = (_Float16*)alloc((size_t)288 * 512 * 2);
  _Float16* Cs  = (_Float16*)alloc((size_t)288 * 512 * 2);
  _Float16* Irm = (_Float16*)alloc((size_t)512 * 288 * 2);
  _Float16* Iim = (_Float16*)alloc((size_t)512 * 288 * 2);

  _Float16* xh[6];
  for (int c = 0; c < 3; ++c)
    for (int p = 0; p < 2; ++p)
      xh[2 * c + p] = (_Float16*)alloc((size_t)Bb * Ll * Mm * ds[c] * 2);

  _Float16 *w1h[3][2], *w2h[3][2], *W1c[3][2], *W2c[3][2];
  float *b1c[3][2], *b2c[3][2];
  for (int c = 0; c < 3; ++c) {
    int d = ds[c];
    for (int p = 0; p < 2; ++p) {
      w1h[c][p] = (_Float16*)alloc((size_t)d * 256 * 2);
      w2h[c][p] = (_Float16*)alloc((size_t)256 * d * 2);
      W1c[c][p] = (_Float16*)alloc((size_t)2 * d * 512 * 2);
      W2c[c][p] = (_Float16*)alloc((size_t)512 * 2 * d * 2);
      b1c[c][p] = (float*)alloc((size_t)512 * 4);
      b2c[c][p] = (float*)alloc((size_t)2 * d * 4);
    }
  }
  _Float16* Xb = (_Float16*)alloc((size_t)Bb * Fp * 2 * Mm * 7 * 2);
  _Float16* Yb = (_Float16*)alloc((size_t)Bb * Fp * 2 * Mm * 7 * 2);
  _Float16* Hb = (_Float16*)alloc((size_t)15840 * 512 * 2);  // reused hidden buffer

  auto P = [&](int c, int k) -> const float* { return (const float*)d_in[6 + c * 24 + k]; };
  auto gsz = [](int n) { return dim3((unsigned)((n + 255) / 256)); };

  // --- setup: twiddles, f16 inputs, packed weights ---
  gen_dft<<<576, 256, 0, stream>>>(Cc, Cs, Irm, Iim);
  for (int k = 0; k < 6; ++k) {
    int n = Bb * Ll * Mm * ds[k >> 1];
    cvt_f32_f16<<<gsz(n), 256, 0, stream>>>((const float*)d_in[k], xh[k], n);
  }
  for (int c = 0; c < 3; ++c) {
    int d = ds[c];
    for (int p = 0; p < 2; ++p) {
      int kw1 = p ? 4 : 0, kw2 = p ? 6 : 2;
      cvt_f32_f16<<<gsz(d * 256), 256, 0, stream>>>(P(c, kw1), w1h[c][p], d * 256);
      cvt_f32_f16<<<gsz(256 * d), 256, 0, stream>>>(P(c, kw2), w2h[c][p], 256 * d);
      int kr = p ? 16 : 8;
      build_w1c<<<gsz(2 * d * 512), 256, 0, stream>>>(W1c[c][p], P(c, kr), P(c, kr + 1), d);
      build_w2c<<<gsz(512 * 2 * d), 256, 0, stream>>>(W2c[c][p], P(c, kr + 4), P(c, kr + 5), d);
      cat2_f32<<<gsz(512), 256, 0, stream>>>(b1c[c][p], P(c, kr + 2), P(c, kr + 3), 256);
      cat2_f32<<<gsz(2 * d), 256, 0, stream>>>(b2c[c][p], P(c, kr + 6), P(c, kr + 7), d);
    }
  }

  auto gemm = [&](const _Float16* A, int lda, int aMode, int aM, int aD,
                  const _Float16* Bm, int ldb,
                  const _Float16* A2, const _Float16* B2,
                  const float* bias, float* oF, _Float16* oH, int ldc,
                  int op, int st, int sM, int sD, float scale,
                  int Mr, int Nn, int Kk) {
    dim3 grid((unsigned)((Nn + 63) / 64), (unsigned)((Mr + 63) / 64));
    gemm_wmma<<<grid, 128, 0, stream>>>(A, lda, aMode, aM, aD, Bm, ldb, A2, B2,
                                        bias, oF, oH, ldc, op, st, sM, sD, scale,
                                        Mr, Nn, Kk);
  };

  float* out = (float*)d_out;
  const long S0 = 112640L * 3, S1 = 112640L * 5, S2 = 112640L * 7;
  const long chOff[3] = {0, S0, S0 + S1};
  const long G = S0 + S1 + S2;  // per-group size

  for (int c = 0; c < 3; ++c) {
    int d = ds[c];
    long LMd = (long)Ll * Mm * d;          // per-batch input/output elements
    int Nd = Mm * d;                        // frequency-domain column count
    long Xstr = (long)Fp * 2 * Mm * d;      // per-batch halves in X/Y buffers
    for (int p = 0; p < 2; ++p) {
      const _Float16* x = xh[2 * c + p];

      // ---- MLP path: relu(X@W1+b1)@W2+b2 ----
      float* outM = out + (long)p * G + chOff[c];
      for (int b = 0; b < 4; ++b) {
        gemm(x + b * LMd, d, 0, 0, 0, w1h[c][p], 256, nullptr, nullptr,
             p ? P(c, 5) : P(c, 1), nullptr, Hb, 256, 2, 1, 0, 0, 1.f,
             Ll * Mm, 256, d);
        gemm(Hb, 256, 0, 0, 0, w2h[c][p], d, nullptr, nullptr,
             p ? P(c, 7) : P(c, 3), outM + b * LMd, nullptr, d, 1, 0, 0, 0, 1.f,
             Ll * Mm, d, 256);
      }

      // ---- temporal path ----
      // T1: rfft as GEMM -> X[b][f][{re,im}][m*d]
      for (int b = 0; b < 4; ++b) {
        gemm(Cc, 512, 0, 0, 0, x + b * LMd, Nd, nullptr, nullptr,
             nullptr, nullptr, Xb + b * Xstr, 2 * Nd, 0, 1, 0, 0, 1.f,
             Fp, Nd, 512);
        gemm(Cs, 512, 0, 0, 0, x + b * LMd, Nd, nullptr, nullptr,
             nullptr, nullptr, Xb + b * Xstr + Nd, 2 * Nd, 0, 1, 0, 0, 1.f,
             Fp, Nd, 512);
      }
      // T2: FreMLP-1 (rows=(f,m), K=2d, N=512) + relu+shrink+dropout-scale -> Hb
      // T3: FreMLP-2 (K=512, N=2d) + relu+shrink, scatter to Y[b][f][{re,im}][m*d]
      for (int b = 0; b < 4; ++b) {
        gemm(Xb + b * Xstr, 0, 1, Mm, d, W1c[c][p], 512, nullptr, nullptr,
             b1c[c][p], nullptr, Hb, 512, 3, 1, 0, 0, sdrop,
             Fp * Mm, 512, 2 * d);
        gemm(Hb, 512, 0, 0, 0, W2c[c][p], 2 * d, nullptr, nullptr,
             b2c[c][p], nullptr, Yb + b * Xstr, 0, 3, 2, Mm, d, 1.f,
             Fp * Mm, 2 * d, 512);
      }
      // T4: irfft as dual GEMM: out = ICr@Yr + ICi@Yi
      float* outT = out + (long)(2 + p) * G + chOff[c];
      for (int b = 0; b < 4; ++b) {
        gemm(Irm, 288, 0, 0, 0, Yb + b * Xstr, 2 * Nd,
             Iim, Yb + b * Xstr + Nd,
             nullptr, outT + b * LMd, nullptr, Nd, 0, 0, 0, 0, 1.f,
             Ll, Nd, 288);
      }
    }
  }
}
